// Adapter_86646670230187
// MI455X (gfx1250) — matrix operand order repaired and checked by reading, not compiled, not run on hardware
//
#include <hip/hip_runtime.h>

typedef __attribute__((ext_vector_type(16))) _Float16 v16h;
typedef __attribute__((ext_vector_type(8)))  _Float16 v8h;
typedef __attribute__((ext_vector_type(8)))  float    v8f;

#define D_MODEL 768
#define BNECK   64
#define TOK_PER_BLOCK 64
#define THREADS 128
#define LN_EPS  1e-5f
#define SCALE   1.0f

// Convert a float4 into 4 consecutive f16 lanes of a v16h
#define CVTQ(dst, base, v)                      \
    dst[(base)+0] = (_Float16)(v).x;            \
    dst[(base)+1] = (_Float16)(v).y;            \
    dst[(base)+2] = (_Float16)(v).z;            \
    dst[(base)+3] = (_Float16)(v).w;

// LayerNorm-fused conversion: ((x-mu)*rs)*g + b -> f16
#define LNQ(dst, base, xv, gv, bv)                                        \
    dst[(base)+0] = (_Float16)(((xv).x - mu_l) * rs_l * (gv).x + (bv).x); \
    dst[(base)+1] = (_Float16)(((xv).y - mu_l) * rs_l * (gv).y + (bv).y); \
    dst[(base)+2] = (_Float16)(((xv).z - mu_l) * rs_l * (gv).z + (bv).z); \
    dst[(base)+3] = (_Float16)(((xv).w - mu_l) * rs_l * (gv).w + (bv).w);

// One-shot weight conversion f32 -> f16 into workspace (shared by all blocks)
__global__ __launch_bounds__(256)
void cvt_weights_kernel(const float* __restrict__ wd, const float* __restrict__ wu,
                        _Float16* __restrict__ wd16, _Float16* __restrict__ wu16)
{
    const int i = blockIdx.x * 256 + threadIdx.x;
    if (i < BNECK * D_MODEL) {
        wd16[i] = (_Float16)wd[i];
        wu16[i] = (_Float16)wu[i];   // both matrices are 64*768 elements
    }
}

template<bool F16W>
__global__ __launch_bounds__(THREADS)
void adapter_fused_kernel(const float* __restrict__ x,
                          const float* __restrict__ ln_w,
                          const float* __restrict__ ln_b,
                          const float* __restrict__ w_down,
                          const float* __restrict__ b_down,
                          const float* __restrict__ w_up,
                          const float* __restrict__ b_up,
                          const _Float16* __restrict__ wd16,
                          const _Float16* __restrict__ wu16,
                          float* __restrict__ out)
{
    // Cross-lane transpose tile: each wave's 16x64 ReLU(down) result.
    // Row stride 72 f16 = 144B (multiple of 16B -> aligned ds_load_b128).
    __shared__ __align__(16) _Float16 sdown[TOK_PER_BLOCK][72];

    const int lane  = threadIdx.x & 31;
    const int wave  = threadIdx.x >> 5;
    const int hi16  = lane >> 4;        // 0 for lanes 0-15, 1 for lanes 16-31
    const int mrow  = lane & 15;        // A-matrix row / B,C,D column within a 16-tile
    const int tokBase = blockIdx.x * TOK_PER_BLOCK + wave * 16;

    // ---------------- LayerNorm statistics (per wave: its 16 tokens) -------
    float mu_l = 0.0f, rs_l = 0.0f;     // stats of token (tokBase + mrow)
    for (int t = 0; t < 16; ++t) {
        const float4* row =
            reinterpret_cast<const float4*>(x + (size_t)(tokBase + t) * D_MODEL);
        float s = 0.0f, s2 = 0.0f;
        #pragma unroll
        for (int j = 0; j < 6; ++j) {          // 32 lanes * 6 float4 = 768
            float4 v = row[lane + j * 32];
            s  += v.x + v.y + v.z + v.w;
            s2 += v.x * v.x + v.y * v.y + v.z * v.z + v.w * v.w;
        }
        #pragma unroll
        for (int off = 16; off >= 1; off >>= 1) {
            s  += __shfl_xor(s,  off, 32);
            s2 += __shfl_xor(s2, off, 32);
        }
        float mu  = s * (1.0f / D_MODEL);
        float var = s2 * (1.0f / D_MODEL) - mu * mu;
        float rs  = __frsqrt_rn(var + LN_EPS);
        if (mrow == t) { mu_l = mu; rs_l = rs; }
    }

    // ---------------- GEMM1: down = relu(xn @ w_down^T + b_down) ----------
    // M=16 tokens (this wave), N=64 (4 tiles), K=768 (24 steps of 32).
    const float* xrow = x + (size_t)(tokBase + mrow) * D_MODEL;

    v8f acc[4] = {};
    for (int kk = 0; kk < 24; ++kk) {
        // A fragment (contiguous k order, matched by B): lane holds row mrow,
        // 16 contiguous K values [k0..k0+15], k0 = kk*32 + 16*hi16.
        const int k0 = kk * 32 + hi16 * 16;
        float4 xa0 = *reinterpret_cast<const float4*>(xrow + k0);
        float4 xa1 = *reinterpret_cast<const float4*>(xrow + k0 + 4);
        float4 xa2 = *reinterpret_cast<const float4*>(xrow + k0 + 8);
        float4 xa3 = *reinterpret_cast<const float4*>(xrow + k0 + 12);
        float4 gw0 = *reinterpret_cast<const float4*>(ln_w + k0);
        float4 gw1 = *reinterpret_cast<const float4*>(ln_w + k0 + 4);
        float4 gw2 = *reinterpret_cast<const float4*>(ln_w + k0 + 8);
        float4 gw3 = *reinterpret_cast<const float4*>(ln_w + k0 + 12);
        float4 gb0 = *reinterpret_cast<const float4*>(ln_b + k0);
        float4 gb1 = *reinterpret_cast<const float4*>(ln_b + k0 + 4);
        float4 gb2 = *reinterpret_cast<const float4*>(ln_b + k0 + 8);
        float4 gb3 = *reinterpret_cast<const float4*>(ln_b + k0 + 12);
        v16h a;
        LNQ(a, 0,  xa0, gw0, gb0)
        LNQ(a, 4,  xa1, gw1, gb1)
        LNQ(a, 8,  xa2, gw2, gb2)
        LNQ(a, 12, xa3, gw3, gb3)

        // B fragment (same contiguous k order as A): lane holds column n = nt*16+mrow,
        // 16 contiguous K values starting at kb = kk*32 + 16*hi16.
        const int kb = kk * 32 + hi16 * 16;
        #pragma unroll
        for (int nt = 0; nt < 4; ++nt) {
            v16h bf;
            if constexpr (F16W) {
                bf = *reinterpret_cast<const v16h*>(
                    wd16 + (size_t)(nt * 16 + mrow) * D_MODEL + kb);
            } else {
                const float* wr = w_down + (size_t)(nt * 16 + mrow) * D_MODEL + kb;
                float4 b0 = *reinterpret_cast<const float4*>(wr);
                float4 b1 = *reinterpret_cast<const float4*>(wr + 4);
                float4 b2 = *reinterpret_cast<const float4*>(wr + 8);
                float4 b3 = *reinterpret_cast<const float4*>(wr + 12);
                CVTQ(bf, 0,  b0)
                CVTQ(bf, 4,  b1)
                CVTQ(bf, 8,  b2)
                CVTQ(bf, 12, b3)
            }
            acc[nt] = __builtin_amdgcn_wmma_f32_16x16x32_f16(
                false, a, false, bf, (short)0, acc[nt], false, false);
        }
    }

    // bias + ReLU, scatter C-layout (lane=col, vgpr=row) into LDS row-major f16
    #pragma unroll
    for (int nt = 0; nt < 4; ++nt) {
        const float bd = b_down[nt * 16 + mrow];
        #pragma unroll
        for (int v = 0; v < 8; ++v) {
            float d = acc[nt][v] + bd;
            d = d > 0.0f ? d : 0.0f;
            sdown[wave * 16 + v + hi16 * 8][nt * 16 + mrow] = (_Float16)d;
        }
    }
    __syncthreads();

    // ---------------- GEMM2: out = down @ w_up^T + b_up + x ---------------
    // M=16 tokens, N=768 (48 tiles), K=64 (2 steps of 32).
    // A2 fragments read back in A-layout via aligned b128 LDS loads.
    v16h a2[2];
    #pragma unroll
    for (int kk = 0; kk < 2; ++kk) {
        const int k0 = kk * 32 + hi16 * 16;
        const _Float16* dr = &sdown[wave * 16 + mrow][0];
        v8h lo = *reinterpret_cast<const v8h*>(dr + k0);
        v8h hh = *reinterpret_cast<const v8h*>(dr + k0 + 8);
        #pragma unroll
        for (int e = 0; e < 8; ++e) { a2[kk][e] = lo[e]; a2[kk][e + 8] = hh[e]; }
    }

    for (int nt = 0; nt < 48; ++nt) {
        const int nd = nt * 16 + mrow;            // d_model column
        v8f c = {};
        #pragma unroll
        for (int kk = 0; kk < 2; ++kk) {
            const int kb = kk * 32 + hi16 * 16;
            v16h bf;
            if constexpr (F16W) {
                bf = *reinterpret_cast<const v16h*>(
                    wu16 + (size_t)nd * BNECK + kb);
            } else {
                const float* wr = w_up + (size_t)nd * BNECK + kb;
                float4 b0 = *reinterpret_cast<const float4*>(wr);
                float4 b1 = *reinterpret_cast<const float4*>(wr + 4);
                float4 b2 = *reinterpret_cast<const float4*>(wr + 8);
                float4 b3 = *reinterpret_cast<const float4*>(wr + 12);
                CVTQ(bf, 0,  b0)
                CVTQ(bf, 4,  b1)
                CVTQ(bf, 8,  b2)
                CVTQ(bf, 12, b3)
            }
            c = __builtin_amdgcn_wmma_f32_16x16x32_f16(
                false, a2[kk], false, bf, (short)0, c, false, false);
        }
        const float bu = b_up[nd];
        #pragma unroll
        for (int v = 0; v < 8; ++v) {
            const size_t idx =
                (size_t)(tokBase + v + hi16 * 8) * D_MODEL + nd;
            // last use of x: non-temporal read; out is write-once: NT store
            const float r = __builtin_nontemporal_load(x + idx);
            __builtin_nontemporal_store((c[v] + bu) * SCALE + r, out + idx);
        }
    }
}

extern "C" void kernel_launch(void* const* d_in, const int* in_sizes, int n_in,
                              void* d_out, int out_size, void* d_ws, size_t ws_size,
                              hipStream_t stream) {
    (void)n_in; (void)out_size;
    const float* x      = (const float*)d_in[0];
    const float* ln_w   = (const float*)d_in[1];
    const float* ln_b   = (const float*)d_in[2];
    const float* w_down = (const float*)d_in[3];
    const float* b_down = (const float*)d_in[4];
    const float* w_up   = (const float*)d_in[5];
    const float* b_up   = (const float*)d_in[6];
    float* out = (float*)d_out;

    const int tokens = in_sizes[0] / D_MODEL;       // 32768
    const int blocks = tokens / TOK_PER_BLOCK;      // 512

    const size_t wbytes = (size_t)2 * BNECK * D_MODEL * sizeof(_Float16); // 196608
    if (ws_size >= wbytes) {
        _Float16* wd16 = reinterpret_cast<_Float16*>(d_ws);
        _Float16* wu16 = wd16 + (size_t)BNECK * D_MODEL;
        cvt_weights_kernel<<<(BNECK * D_MODEL + 255) / 256, 256, 0, stream>>>(
            w_down, w_up, wd16, wu16);
        adapter_fused_kernel<true><<<blocks, THREADS, 0, stream>>>(
            x, ln_w, ln_b, w_down, b_down, w_up, b_up, wd16, wu16, out);
    } else {
        adapter_fused_kernel<false><<<blocks, THREADS, 0, stream>>>(
            x, ln_w, ln_b, w_down, b_down, w_up, b_up, nullptr, nullptr, out);
    }
}
